// HierarchicalBlock_4629974745487
// MI455X (gfx1250) — matrix operand order repaired and checked by reading, not compiled, not run on hardware
//
#include <hip/hip_runtime.h>

// ---------------- problem constants ----------------
#define DMODEL 512
#define NHEADS 8
#define HDIM   64
#define WINSZ  128
#define SEQ    4096
#define BATCHN 4
#define DFFN   1536
#define MROWS  (BATCHN*SEQ)        // 16384
#define BHN    (BATCHN*NHEADS)     // 32
#define EPSV   1e-6f
#define ATT_SCALE 0.125f           // HDIM^-0.5

// ---------------- types ----------------
typedef __attribute__((ext_vector_type(16))) __bf16 v16bf;
typedef __attribute__((ext_vector_type(8)))  __bf16 v8bf;
typedef __attribute__((ext_vector_type(8)))  float  v8f;

__device__ __forceinline__ unsigned short f32_to_bf16(float f) {
    unsigned int u = __float_as_uint(f);
    u += 0x7FFFu + ((u >> 16) & 1u);               // round-to-nearest-even
    return (unsigned short)(u >> 16);
}
__device__ __forceinline__ float bf16_to_f32(unsigned short h) {
    return __uint_as_float(((unsigned int)h) << 16);
}

// A-operand fragment (16xK row-major panel): lane holds row (lane&15),
// K chunks [hi*8, hi*8+8) and [16+hi*8, 24+hi*8)   (ISA 7.12.2, 16-bit A 16x32)
__device__ __forceinline__ v16bf load_a_frag(const unsigned short* p, int ld) {
    const int lane = threadIdx.x & 31;
    const int i  = lane & 15;
    const int hi = lane >> 4;
    const unsigned short* r = p + (size_t)i * ld + hi * 8;
    v8bf a = *reinterpret_cast<const v8bf*>(r);
    v8bf b = *reinterpret_cast<const v8bf*>(r + 16);
    v16bf o;
#pragma unroll
    for (int t = 0; t < 8; ++t) { o[t] = a[t]; o[8 + t] = b[t]; }
    return o;
}

// B-operand fragment from an N x K row-major panel (i.e. B^T): lane holds
// column (lane&15), K chunks [hi*8, hi*8+8) and [16+hi*8, 24+hi*8)
__device__ __forceinline__ v16bf load_b_frag(const unsigned short* p, int ld) {
    const int lane = threadIdx.x & 31;
    const int n  = lane & 15;
    const int hi = lane >> 4;
    const unsigned short* r = p + (size_t)n * ld + hi * 8;
    v8bf a = *reinterpret_cast<const v8bf*>(r);
    v8bf b = *reinterpret_cast<const v8bf*>(r + 16);
    v16bf o;
#pragma unroll
    for (int t = 0; t < 8; ++t) { o[t] = a[t]; o[8 + t] = b[t]; }
    return o;
}

#define WMMA_BF16(A,B,C) __builtin_amdgcn_wmma_f32_16x16x32_bf16(false,(A),false,(B),(short)0,(C),false,false)

// ---------------- small utility kernels ----------------
__global__ void conv_transpose_kernel(const float* __restrict__ W,
                                      unsigned short* __restrict__ Wt,
                                      int K, int N) {
    size_t gid = (size_t)blockIdx.x * 256 + threadIdx.x;
    if (gid >= (size_t)K * N) return;
    int k = (int)(gid / N), n = (int)(gid % N);
    Wt[(size_t)n * K + k] = f32_to_bf16(W[gid]);
}

__global__ __launch_bounds__(128)
void rmsnorm_bf16_kernel(const float* __restrict__ x, const float* __restrict__ w,
                         unsigned short* __restrict__ out) {
    const int row = blockIdx.x;
    const int tid = threadIdx.x;          // 128 threads, 4 elems each
    const float4 xv = reinterpret_cast<const float4*>(x + (size_t)row * DMODEL)[tid];
    float ss = xv.x*xv.x + xv.y*xv.y + xv.z*xv.z + xv.w*xv.w;
#pragma unroll
    for (int m = 1; m < 32; m <<= 1) ss += __shfl_xor(ss, m, 32);
    __shared__ float red[4];
    if ((tid & 31) == 0) red[tid >> 5] = ss;
    __syncthreads();
    ss = red[0] + red[1] + red[2] + red[3];
    const float sc = rsqrtf(ss * (1.0f / DMODEL) + EPSV);
    const float4 wv = reinterpret_cast<const float4*>(w)[tid];
    unsigned long long pk =
        (unsigned long long)f32_to_bf16(xv.x * sc * wv.x)
      | ((unsigned long long)f32_to_bf16(xv.y * sc * wv.y) << 16)
      | ((unsigned long long)f32_to_bf16(xv.z * sc * wv.z) << 32)
      | ((unsigned long long)f32_to_bf16(xv.w * sc * wv.w) << 48);
    reinterpret_cast<unsigned long long*>(out + (size_t)row * DMODEL)[tid] = pk;
}

// (M,D) f32 ->  bf16 (b,H,s,hd)
__global__ void pack_qk_kernel(const float* __restrict__ in, unsigned short* __restrict__ out) {
    size_t gid = (size_t)blockIdx.x * 256 + threadIdx.x;
    if (gid >= (size_t)MROWS * DMODEL) return;
    int m = (int)(gid / DMODEL), c = (int)(gid % DMODEL);
    int b = m / SEQ, s = m % SEQ, h = c / HDIM, d = c % HDIM;
    out[(((size_t)(b * NHEADS + h)) * SEQ + s) * HDIM + d] = f32_to_bf16(in[gid]);
}
// (M,D) f32 ->  bf16 (b,H,hd,s)   (transposed V for P*V operand)
__global__ void pack_vT_kernel(const float* __restrict__ in, unsigned short* __restrict__ out) {
    size_t gid = (size_t)blockIdx.x * 256 + threadIdx.x;
    if (gid >= (size_t)MROWS * DMODEL) return;
    int m = (int)(gid / DMODEL), c = (int)(gid % DMODEL);
    int b = m / SEQ, s = m % SEQ, h = c / HDIM, d = c % HDIM;
    out[(((size_t)(b * NHEADS + h)) * HDIM + d) * SEQ + s] = f32_to_bf16(in[gid]);
}
// (b,H,s,hd) f32 -> bf16 (M, D)
__global__ void pack_attn_kernel(const float* __restrict__ in, unsigned short* __restrict__ out) {
    size_t gid = (size_t)blockIdx.x * 256 + threadIdx.x;
    if (gid >= (size_t)MROWS * DMODEL) return;
    int bh = (int)(gid / ((size_t)SEQ * HDIM));
    int r  = (int)(gid % ((size_t)SEQ * HDIM));
    int s = r / HDIM, d = r % HDIM;
    int b = bh / NHEADS, h = bh % NHEADS;
    out[((size_t)(b * SEQ + s)) * DMODEL + h * HDIM + d] = f32_to_bf16(in[gid]);
}

// mean-pool K (b,H,s,hd)->(b,H,nc,hd) and Vt (b,H,hd,s)->(b,H,hd,nc)
__global__ void pool_kv_kernel(const unsigned short* __restrict__ kb,
                               const unsigned short* __restrict__ vT,
                               unsigned short* __restrict__ kc,
                               unsigned short* __restrict__ vcT,
                               int C, int nc) {
    int gid = blockIdx.x * 256 + threadIdx.x;
    int total = BHN * nc * HDIM;
    if (gid >= 2 * total) return;
    const float inv = 1.0f / (float)C;
    if (gid < total) {
        int bh = gid / (nc * HDIM); int r = gid % (nc * HDIM);
        int c = r / HDIM, d = r % HDIM;
        const unsigned short* p = kb + ((size_t)bh * SEQ + (size_t)c * C) * HDIM + d;
        float s = 0.f;
        for (int j = 0; j < C; ++j) s += bf16_to_f32(p[(size_t)j * HDIM]);
        kc[((size_t)bh * nc + c) * HDIM + d] = f32_to_bf16(s * inv);
    } else {
        int g = gid - total;
        int bh = g / (nc * HDIM); int r = g % (nc * HDIM);
        int d = r / nc, c = r % nc;
        const unsigned short* p = vT + ((size_t)bh * HDIM + d) * SEQ + (size_t)c * C;
        float s = 0.f;
        for (int j = 0; j < C; ++j) s += bf16_to_f32(p[j]);
        vcT[((size_t)bh * HDIM + d) * nc + c] = f32_to_bf16(s * inv);
    }
}

// ---------------- WMMA GEMM: C(MxN) = A(MxK) * Bt(NxK)^T ----------------
// Block = 128 threads (4 waves) computing a 64x64 tile. A and B(±B2) panels
// (64x32 bf16 = 4 KB each) are staged into double-buffered LDS with the CDNA5
// async DMA path (GLOBAL_LOAD_ASYNC_TO_LDS_B128, tracked by ASYNCcnt) so the
// copy of k-chunk c+1 overlaps the WMMAs of chunk c.
enum { GM_F32 = 0, GM_F32_RES = 1, GM_SWIGLU = 2 };

template<int MODE>
__global__ __launch_bounds__(128)
void gemm_wmma_kernel(const unsigned short* __restrict__ A,
                      const unsigned short* __restrict__ Bt,
                      const unsigned short* __restrict__ Bt2,
                      const float* __restrict__ res,
                      float* __restrict__ Cf,
                      unsigned short* __restrict__ Cb,
                      int M, int N, int K) {
    const int tid  = threadIdx.x;
    const int wid  = tid >> 5;
    const int lane = tid & 31;
    const int lo = lane & 15, hi = lane >> 4;
    const int bnb = blockIdx.x * 64;
    const int bmb = blockIdx.y * 64;
    const int wn = (wid & 1) * 32;
    const int wm = (wid >> 1) * 32;

    __shared__ __attribute__((aligned(16))) unsigned short Ash[2][64][32];
    __shared__ __attribute__((aligned(16))) unsigned short Bsh[2][64][32];
    __shared__ __attribute__((aligned(16))) unsigned short B2sh[(MODE == GM_SWIGLU) ? 2 : 1][64][32];

    const unsigned long long abase  = (unsigned long long)(size_t)A;
    const unsigned long long bbase  = (unsigned long long)(size_t)Bt;
    const unsigned long long b2base = (unsigned long long)(size_t)Bt2;

    // issue one 64x32 A panel + 64x32 B panel (+B2) for k-chunk kc into LDS
    auto issue = [&](int kc) {
        const int k0  = kc << 5;
        const int buf = kc & 1;
#pragma unroll
        for (int half = 0; half < 2; ++half) {
            const int u   = tid + half * 128;      // 256 x 16B units per panel
            const int row = u >> 2;
            const int ks  = (u & 3) * 8;
            const unsigned aoff = (unsigned)(((bmb + row) * K + k0 + ks) * 2);
            const unsigned alds = (unsigned)(size_t)&Ash[buf][row][ks];
            asm volatile("global_load_async_to_lds_b128 %0, %1, %2 offset:0"
                         :: "v"(alds), "v"(aoff), "s"(abase) : "memory");
            const unsigned boff = (unsigned)(((bnb + row) * K + k0 + ks) * 2);
            const unsigned blds = (unsigned)(size_t)&Bsh[buf][row][ks];
            asm volatile("global_load_async_to_lds_b128 %0, %1, %2 offset:0"
                         :: "v"(blds), "v"(boff), "s"(bbase) : "memory");
            if (MODE == GM_SWIGLU) {
                const unsigned b2lds = (unsigned)(size_t)&B2sh[buf][row][ks];
                asm volatile("global_load_async_to_lds_b128 %0, %1, %2 offset:0"
                             :: "v"(b2lds), "v"(boff), "s"(b2base) : "memory");
            }
        }
    };

    v8f acc[2][2]  = {};
    v8f acc2[2][2] = {};
    const int nK = K >> 5;

    issue(0);
    for (int kc = 0; kc < nK; ++kc) {
        const int buf = kc & 1;
        if (kc + 1 < nK) {
            issue(kc + 1);
            // one chunk (4 or 6 per-wave async ops) may remain in flight;
            // async loads complete in order, so waiting <=N drains chunk kc.
            if (MODE == GM_SWIGLU) asm volatile("s_wait_asynccnt 0x6" ::: "memory");
            else                   asm volatile("s_wait_asynccnt 0x4" ::: "memory");
        } else {
            asm volatile("s_wait_asynccnt 0x0" ::: "memory");
        }
        __syncthreads();                           // panel visible to all waves
        v16bf a0 = load_a_frag(&Ash[buf][wm][0], 32);
        v16bf a1 = load_a_frag(&Ash[buf][wm + 16][0], 32);
        v16bf b0 = load_b_frag(&Bsh[buf][wn][0], 32);
        v16bf b1 = load_b_frag(&Bsh[buf][wn + 16][0], 32);
        acc[0][0] = WMMA_BF16(a0, b0, acc[0][0]);
        acc[0][1] = WMMA_BF16(a0, b1, acc[0][1]);
        acc[1][0] = WMMA_BF16(a1, b0, acc[1][0]);
        acc[1][1] = WMMA_BF16(a1, b1, acc[1][1]);
        if (MODE == GM_SWIGLU) {
            v16bf c0 = load_b_frag(&B2sh[buf][wn][0], 32);
            v16bf c1 = load_b_frag(&B2sh[buf][wn + 16][0], 32);
            acc2[0][0] = WMMA_BF16(a0, c0, acc2[0][0]);
            acc2[0][1] = WMMA_BF16(a0, c1, acc2[0][1]);
            acc2[1][0] = WMMA_BF16(a1, c0, acc2[1][0]);
            acc2[1][1] = WMMA_BF16(a1, c1, acc2[1][1]);
        }
        __syncthreads();                           // all reads done before buffer reuse
    }
#pragma unroll
    for (int mi = 0; mi < 2; ++mi)
#pragma unroll
        for (int ni = 0; ni < 2; ++ni) {
            const int mrow = bmb + wm + mi * 16 + hi * 8;
            const int ncol = bnb + wn + ni * 16 + lo;
#pragma unroll
            for (int v = 0; v < 8; ++v) {
                size_t idx = (size_t)(mrow + v) * N + ncol;
                float xv = acc[mi][ni][v];
                if (MODE == GM_F32) {
                    Cf[idx] = xv;
                } else if (MODE == GM_F32_RES) {
                    Cf[idx] = res[idx] + xv;
                } else {
                    float g = xv / (1.0f + __expf(-xv));          // silu
                    Cb[idx] = f32_to_bf16(g * acc2[mi][ni][v]);
                }
            }
        }
}

// ---------------- level 0: causal local-window flash attention ----------------
__global__ __launch_bounds__(256)
void local_attn_kernel(const unsigned short* __restrict__ qb,
                       const unsigned short* __restrict__ kb,
                       const unsigned short* __restrict__ vT,
                       float* __restrict__ out) {
    const int bh  = blockIdx.z * NHEADS + blockIdx.y;
    const int qtb = blockIdx.x * WINSZ;              // block query base
    const int wid = threadIdx.x >> 5;
    const int lane = threadIdx.x & 31;
    const int lo = lane & 15, hi = lane >> 4;
    const int qt = qtb + wid * 16;                   // wave's 16 query rows

    const unsigned short* qrow = qb + ((size_t)bh * SEQ + qt) * HDIM;
    v16bf qa0 = load_a_frag(qrow, HDIM);
    v16bf qa1 = load_a_frag(qrow + 32, HDIM);

    float rmax[8], rsum[8];
    v8f acc[4] = {};
#pragma unroll
    for (int v = 0; v < 8; ++v) { rmax[v] = -3.0e38f; rsum[v] = 0.f; }

    __shared__ __attribute__((aligned(16))) unsigned short plds[8][16][32];

    int ks = qtb - WINSZ + 1; ks &= ~31; if (ks < 0) ks = 0;   // block-uniform
    for (int kp0 = ks; kp0 < qtb + WINSZ; kp0 += 32) {
        v8f st[2] = {};
#pragma unroll
        for (int t = 0; t < 2; ++t) {
            const unsigned short* kbase = kb + ((size_t)bh * SEQ + kp0 + t * 16) * HDIM;
            v16bf b0 = load_b_frag(kbase, HDIM);
            v16bf b1 = load_b_frag(kbase + 32, HDIM);
            st[t] = WMMA_BF16(qa0, b0, st[t]);
            st[t] = WMMA_BF16(qa1, b1, st[t]);
        }
        float pv[2][8], cm[8];
#pragma unroll
        for (int v = 0; v < 8; ++v) cm[v] = -3.0e38f;
#pragma unroll
        for (int t = 0; t < 2; ++t)
#pragma unroll
            for (int v = 0; v < 8; ++v) {
                int row = qt + hi * 8 + v;
                int col = kp0 + t * 16 + lo;
                float s = st[t][v] * ATT_SCALE;
                if (col <= row - WINSZ || col > row) s = -1e30f;
                pv[t][v] = s;
                cm[v] = fmaxf(cm[v], s);
            }
#pragma unroll
        for (int m = 1; m < 16; m <<= 1)
#pragma unroll
            for (int v = 0; v < 8; ++v) cm[v] = fmaxf(cm[v], __shfl_xor(cm[v], m, 32));
        float ef[8], ps[8];
#pragma unroll
        for (int v = 0; v < 8; ++v) {
            float nm = fmaxf(rmax[v], cm[v]);
            ef[v] = __expf(rmax[v] - nm);
            rmax[v] = nm; ps[v] = 0.f;
        }
#pragma unroll
        for (int t = 0; t < 2; ++t)
#pragma unroll
            for (int v = 0; v < 8; ++v) { float p = __expf(pv[t][v] - rmax[v]); pv[t][v] = p; ps[v] += p; }
#pragma unroll
        for (int m = 1; m < 16; m <<= 1)
#pragma unroll
            for (int v = 0; v < 8; ++v) ps[v] += __shfl_xor(ps[v], m, 32);
#pragma unroll
        for (int v = 0; v < 8; ++v) rsum[v] = rsum[v] * ef[v] + ps[v];
#pragma unroll
        for (int n = 0; n < 4; ++n)
#pragma unroll
            for (int v = 0; v < 8; ++v) acc[n][v] *= ef[v];
        __syncthreads();
#pragma unroll
        for (int t = 0; t < 2; ++t)
#pragma unroll
            for (int v = 0; v < 8; ++v) plds[wid][hi * 8 + v][t * 16 + lo] = f32_to_bf16(pv[t][v]);
        __syncthreads();
        v16bf pa = load_a_frag(&plds[wid][0][0], 32);
#pragma unroll
        for (int n = 0; n < 4; ++n) {
            v16bf vb = load_b_frag(vT + ((size_t)bh * HDIM + n * 16) * SEQ + kp0, SEQ);
            acc[n] = WMMA_BF16(pa, vb, acc[n]);
        }
    }
#pragma unroll
    for (int n = 0; n < 4; ++n)
#pragma unroll
        for (int v = 0; v < 8; ++v) {
            int row = qt + hi * 8 + v;
            out[((size_t)bh * SEQ + row) * HDIM + n * 16 + lo] = acc[n][v] / rsum[v];
        }
}

// ---------------- levels 1..2: causal attention over pooled chunks ----------------
__global__ __launch_bounds__(256)
void lvl_attn_kernel(const unsigned short* __restrict__ qb,
                     const unsigned short* __restrict__ kc,
                     const unsigned short* __restrict__ vcT,
                     const float* __restrict__ gamma, int gidx,
                     int C, int nc,
                     float* __restrict__ out) {
    const int bh  = blockIdx.z * NHEADS + blockIdx.y;
    const int qtb = blockIdx.x * 128;
    const int wid = threadIdx.x >> 5;
    const int lane = threadIdx.x & 31;
    const int lo = lane & 15, hi = lane >> 4;
    const int qt = qtb + wid * 16;

    const unsigned short* qrow = qb + ((size_t)bh * SEQ + qt) * HDIM;
    v16bf qa0 = load_a_frag(qrow, HDIM);
    v16bf qa1 = load_a_frag(qrow + 32, HDIM);

    float rmax[8], rsum[8];
    v8f acc[4] = {};
#pragma unroll
    for (int v = 0; v < 8; ++v) { rmax[v] = -3.0e38f; rsum[v] = 0.f; }

    __shared__ __attribute__((aligned(16))) unsigned short plds[8][16][32];

    int cmax = (qtb + 127) / C; if (cmax > nc) cmax = nc;      // block-uniform
    for (int c0 = 0; c0 < cmax; c0 += 32) {
        v8f st[2] = {};
#pragma unroll
        for (int t = 0; t < 2; ++t) {
            const unsigned short* kbase = kc + ((size_t)bh * nc + c0 + t * 16) * HDIM;
            v16bf b0 = load_b_frag(kbase, HDIM);
            v16bf b1 = load_b_frag(kbase + 32, HDIM);
            st[t] = WMMA_BF16(qa0, b0, st[t]);
            st[t] = WMMA_BF16(qa1, b1, st[t]);
        }
        float pv[2][8], cm[8];
#pragma unroll
        for (int v = 0; v < 8; ++v) cm[v] = -3.0e38f;
#pragma unroll
        for (int t = 0; t < 2; ++t)
#pragma unroll
            for (int v = 0; v < 8; ++v) {
                int row = qt + hi * 8 + v;
                int col = c0 + t * 16 + lo;
                float s = st[t][v] * ATT_SCALE;
                if (col >= row / C || col >= nc) s = -1e30f;   // chunk strictly in past
                pv[t][v] = s;
                cm[v] = fmaxf(cm[v], s);
            }
#pragma unroll
        for (int m = 1; m < 16; m <<= 1)
#pragma unroll
            for (int v = 0; v < 8; ++v) cm[v] = fmaxf(cm[v], __shfl_xor(cm[v], m, 32));
        float ef[8], ps[8];
#pragma unroll
        for (int v = 0; v < 8; ++v) {
            float nm = fmaxf(rmax[v], cm[v]);
            ef[v] = __expf(rmax[v] - nm);
            rmax[v] = nm; ps[v] = 0.f;
        }
#pragma unroll
        for (int t = 0; t < 2; ++t)
#pragma unroll
            for (int v = 0; v < 8; ++v) { float p = __expf(pv[t][v] - rmax[v]); pv[t][v] = p; ps[v] += p; }
#pragma unroll
        for (int m = 1; m < 16; m <<= 1)
#pragma unroll
            for (int v = 0; v < 8; ++v) ps[v] += __shfl_xor(ps[v], m, 32);
#pragma unroll
        for (int v = 0; v < 8; ++v) rsum[v] = rsum[v] * ef[v] + ps[v];
#pragma unroll
        for (int n = 0; n < 4; ++n)
#pragma unroll
            for (int v = 0; v < 8; ++v) acc[n][v] *= ef[v];
        __syncthreads();
#pragma unroll
        for (int t = 0; t < 2; ++t)
#pragma unroll
            for (int v = 0; v < 8; ++v) plds[wid][hi * 8 + v][t * 16 + lo] = f32_to_bf16(pv[t][v]);
        __syncthreads();
        v16bf pa = load_a_frag(&plds[wid][0][0], 32);
#pragma unroll
        for (int n = 0; n < 4; ++n) {
            v16bf vb = load_b_frag(vcT + ((size_t)bh * HDIM + n * 16) * nc + c0, nc);
            acc[n] = WMMA_BF16(pa, vb, acc[n]);
        }
    }
    const float g = gamma[gidx];
#pragma unroll
    for (int n = 0; n < 4; ++n)
#pragma unroll
        for (int v = 0; v < 8; ++v) {
            int row = qt + hi * 8 + v;
            float den = rsum[v];
            if (row >= C && den > 0.f)
                out[((size_t)bh * SEQ + row) * HDIM + n * 16 + lo] += g * acc[n][v] / den;
        }
}

// ---------------- level 3 (nc = 8): tiny scalar kernel ----------------
__global__ __launch_bounds__(256)
void lvl3_attn_kernel(const unsigned short* __restrict__ qb,
                      const unsigned short* __restrict__ kc,
                      const unsigned short* __restrict__ vcT,
                      const float* __restrict__ gamma,
                      float* __restrict__ out) {
    int gid = blockIdx.x * 256 + threadIdx.x;
    if (gid >= BHN * SEQ) return;
    int bh = gid / SEQ, s = gid % SEQ;
    int nval = s / 512;
    if (nval == 0) return;                  // no valid chunk -> zero contribution
    const unsigned short* qp = qb + ((size_t)bh * SEQ + s) * HDIM;
    float sc[8];
    float mx = -3.0e38f;
    for (int c = 0; c < nval; ++c) {
        const unsigned short* kp = kc + ((size_t)bh * 8 + c) * HDIM;
        float a = 0.f;
        for (int d = 0; d < HDIM; ++d) a += bf16_to_f32(qp[d]) * bf16_to_f32(kp[d]);
        sc[c] = a * ATT_SCALE;
        mx = fmaxf(mx, sc[c]);
    }
    float den = 0.f;
    for (int c = 0; c < nval; ++c) { sc[c] = __expf(sc[c] - mx); den += sc[c]; }
    const float g = gamma[2] / den;
    float* op = out + ((size_t)bh * SEQ + s) * HDIM;
    for (int d = 0; d < HDIM; ++d) {
        const unsigned short* vp = vcT + ((size_t)bh * HDIM + d) * 8;
        float a = 0.f;
        for (int c = 0; c < nval; ++c) a += sc[c] * bf16_to_f32(vp[c]);
        op[d] += g * a;
    }
}

// ---------------- launcher ----------------
extern "C" void kernel_launch(void* const* d_in, const int* in_sizes, int n_in,
                              void* d_out, int out_size, void* d_ws, size_t ws_size,
                              hipStream_t stream) {
    const float* x     = (const float*)d_in[0];
    const float* n1w   = (const float*)d_in[1];
    const float* n2w   = (const float*)d_in[2];
    const float* wq    = (const float*)d_in[3];
    const float* wk    = (const float*)d_in[4];
    const float* wv    = (const float*)d_in[5];
    const float* wo    = (const float*)d_in[6];
    const float* gamma = (const float*)d_in[7];
    const float* w1    = (const float*)d_in[8];
    const float* w2    = (const float*)d_in[9];
    const float* w3    = (const float*)d_in[10];
    float* out = (float*)d_out;

    char* ws = (char*)d_ws;
    size_t off = 0;
    auto alloc = [&](size_t bytes) -> void* {
        void* p = ws + off;
        off = (off + bytes + 255) & ~(size_t)255;
        return p;
    };
    const size_t MD2 = (size_t)MROWS * DMODEL * 2;
    const size_t MD4 = (size_t)MROWS * DMODEL * 4;

    unsigned short* xnb   = (unsigned short*)alloc(MD2);          // rmsnorm out (bf16), reused
    unsigned short* qbuf  = (unsigned short*)alloc(MD2);          // (b,H,s,hd)
    unsigned short* kbuf  = (unsigned short*)alloc(MD2);          // (b,H,s,hd)
    unsigned short* vTbuf = (unsigned short*)alloc(MD2);          // (b,H,hd,s)
    float*          attnf = (float*)alloc(MD4);                   // (b,H,s,hd)
    float*          hbuf  = (float*)alloc(MD4);                   // residual stream
    unsigned short* apack = (unsigned short*)alloc(MD2);          // attn out packed (M,D)
    char*           big   = (char*)alloc((size_t)MROWS * DFFN * 2); // tmp f32 (32MB) / g bf16 (48MB)
    float*          tmpf  = (float*)big;
    unsigned short* gbuf  = (unsigned short*)big;
    unsigned short* wqT = (unsigned short*)alloc((size_t)DMODEL * DMODEL * 2);
    unsigned short* wkT = (unsigned short*)alloc((size_t)DMODEL * DMODEL * 2);
    unsigned short* wvT = (unsigned short*)alloc((size_t)DMODEL * DMODEL * 2);
    unsigned short* woT = (unsigned short*)alloc((size_t)DMODEL * DMODEL * 2);
    unsigned short* w1T = (unsigned short*)alloc((size_t)DMODEL * DFFN * 2);
    unsigned short* w2T = (unsigned short*)alloc((size_t)DMODEL * DFFN * 2);
    unsigned short* w3T = (unsigned short*)alloc((size_t)DFFN * DMODEL * 2);
    unsigned short* kc1 = (unsigned short*)alloc((size_t)BHN * 512 * HDIM * 2);
    unsigned short* vc1 = (unsigned short*)alloc((size_t)BHN * 512 * HDIM * 2);
    unsigned short* kc2 = (unsigned short*)alloc((size_t)BHN * 64 * HDIM * 2);
    unsigned short* vc2 = (unsigned short*)alloc((size_t)BHN * 64 * HDIM * 2);
    unsigned short* kc3 = (unsigned short*)alloc((size_t)BHN * 8 * HDIM * 2);
    unsigned short* vc3 = (unsigned short*)alloc((size_t)BHN * 8 * HDIM * 2);

    auto cdiv = [](size_t a, size_t b) { return (unsigned)((a + b - 1) / b); };

    // 1) weight bf16 transposes
    conv_transpose_kernel<<<cdiv((size_t)DMODEL*DMODEL,256),256,0,stream>>>(wq, wqT, DMODEL, DMODEL);
    conv_transpose_kernel<<<cdiv((size_t)DMODEL*DMODEL,256),256,0,stream>>>(wk, wkT, DMODEL, DMODEL);
    conv_transpose_kernel<<<cdiv((size_t)DMODEL*DMODEL,256),256,0,stream>>>(wv, wvT, DMODEL, DMODEL);
    conv_transpose_kernel<<<cdiv((size_t)DMODEL*DMODEL,256),256,0,stream>>>(wo, woT, DMODEL, DMODEL);
    conv_transpose_kernel<<<cdiv((size_t)DMODEL*DFFN,256),256,0,stream>>>(w1, w1T, DMODEL, DFFN);
    conv_transpose_kernel<<<cdiv((size_t)DMODEL*DFFN,256),256,0,stream>>>(w2, w2T, DMODEL, DFFN);
    conv_transpose_kernel<<<cdiv((size_t)DFFN*DMODEL,256),256,0,stream>>>(w3, w3T, DFFN, DMODEL);

    // 2) rmsnorm(x) -> bf16
    rmsnorm_bf16_kernel<<<MROWS,128,0,stream>>>(x, n1w, xnb);

    // 3) Q,K,V projections (WMMA), repack into attention layouts
    dim3 gemmGrid(DMODEL/64, MROWS/64);
    const unsigned packBlocks = cdiv((size_t)MROWS*DMODEL, 256);
    gemm_wmma_kernel<GM_F32><<<gemmGrid,128,0,stream>>>(xnb, wqT, nullptr, nullptr, tmpf, nullptr, MROWS, DMODEL, DMODEL);
    pack_qk_kernel<<<packBlocks,256,0,stream>>>(tmpf, qbuf);
    gemm_wmma_kernel<GM_F32><<<gemmGrid,128,0,stream>>>(xnb, wkT, nullptr, nullptr, tmpf, nullptr, MROWS, DMODEL, DMODEL);
    pack_qk_kernel<<<packBlocks,256,0,stream>>>(tmpf, kbuf);
    gemm_wmma_kernel<GM_F32><<<gemmGrid,128,0,stream>>>(xnb, wvT, nullptr, nullptr, tmpf, nullptr, MROWS, DMODEL, DMODEL);
    pack_vT_kernel<<<packBlocks,256,0,stream>>>(tmpf, vTbuf);

    // 4) level 0 local-window flash attention (writes attnf)
    local_attn_kernel<<<dim3(SEQ/WINSZ, NHEADS, BATCHN),256,0,stream>>>(qbuf, kbuf, vTbuf, attnf);

    // 5) chunk pooling for levels 1..3
    pool_kv_kernel<<<cdiv((size_t)2*BHN*512*HDIM,256),256,0,stream>>>(kbuf, vTbuf, kc1, vc1, 8, 512);
    pool_kv_kernel<<<cdiv((size_t)2*BHN*64*HDIM,256),256,0,stream>>>(kbuf, vTbuf, kc2, vc2, 64, 64);
    pool_kv_kernel<<<cdiv((size_t)2*BHN*8*HDIM,256),256,0,stream>>>(kbuf, vTbuf, kc3, vc3, 512, 8);

    // 6) hierarchical levels accumulate into attnf
    lvl_attn_kernel<<<dim3(SEQ/128, NHEADS, BATCHN),256,0,stream>>>(qbuf, kc1, vc1, gamma, 0, 8, 512, attnf);
    lvl_attn_kernel<<<dim3(SEQ/128, NHEADS, BATCHN),256,0,stream>>>(qbuf, kc2, vc2, gamma, 1, 64, 64, attnf);
    lvl3_attn_kernel<<<cdiv((size_t)BHN*SEQ,256),256,0,stream>>>(qbuf, kc3, vc3, gamma, attnf);

    // 7) output projection + residual:  h = x + attn @ wo
    pack_attn_kernel<<<packBlocks,256,0,stream>>>(attnf, apack);
    gemm_wmma_kernel<GM_F32_RES><<<gemmGrid,128,0,stream>>>(apack, woT, nullptr, x, hbuf, nullptr, MROWS, DMODEL, DMODEL);

    // 8) FFN: rmsnorm(h) -> fused SwiGLU gate (bf16) -> down-proj + residual
    rmsnorm_bf16_kernel<<<MROWS,128,0,stream>>>(hbuf, n2w, xnb);
    gemm_wmma_kernel<GM_SWIGLU><<<dim3(DFFN/64, MROWS/64),128,0,stream>>>(xnb, w1T, w2T, nullptr, nullptr, gbuf, MROWS, DFFN, DMODEL);
    gemm_wmma_kernel<GM_F32_RES><<<gemmGrid,128,0,stream>>>(gbuf, w3T, nullptr, hbuf, out, nullptr, MROWS, DMODEL, DFFN);

    (void)in_sizes; (void)n_in; (void)out_size; (void)ws_size;
}
